// LocalForwardTemporalAttention_90898687852562
// MI455X (gfx1250) — matrix operand order repaired and checked by reading, not compiled, not run on hardware
//
#include <hip/hip_runtime.h>

// ---------------------------------------------------------------------------
// LocalForwardTemporalAttention for MI455X (gfx1250).
// bf16 WMMA (v_wmma_f32_16x16x32_bf16) for every GEMM, f32 accumulate.
// Global->LDS tile staging via the Tensor Data Mover (tensor_load_to_lds),
// double-buffered, synchronized with s_wait_tensorcnt.
// Shapes: B=8, HW=32 (T=31), NF=256, E=512, M=2048, H=8, dh=64.
// ---------------------------------------------------------------------------

typedef __attribute__((ext_vector_type(16))) __bf16 v16bf;
typedef __attribute__((ext_vector_type(8)))  float  v8f;
typedef __attribute__((ext_vector_type(4)))  unsigned uint32x4;
typedef __attribute__((ext_vector_type(8)))  int      int32x8;
typedef __attribute__((ext_vector_type(4)))  int      int32x4;

union AFrag { v16bf v; unsigned u[8]; };

__device__ inline v8f v8f_zero() {
  v8f z;
#pragma unroll
  for (int i = 0; i < 8; ++i) z[i] = 0.0f;
  return z;
}

__device__ inline v8f wmma_bf16(const AFrag& a, const v16bf& b, v8f c) {
  return __builtin_amdgcn_wmma_f32_16x16x32_bf16(
      false, a.v, false, b, (short)0, c, false, false);
}

// A fragment (16x32 bf16, row-major source, `stride` elems/row).
// This code's k order: lane = M row (lane&15); half-wave g = lane>>4:
//   vgpr v -> K = 16g + 2v (pair)
__device__ inline AFrag load_a(const __bf16* base, int stride, int lane) {
  AFrag f;
  int m = lane & 15, g = lane >> 4;
  const __bf16* rp = base + (size_t)m * stride;
#pragma unroll
  for (int v = 0; v < 8; ++v) {
    int kv = g * 16 + 2 * v;
    f.u[v] = *reinterpret_cast<const unsigned*>(rp + kv);
  }
  return f;
}

// B fragment (32x16 bf16) from N-major source (base + n*stride + k):
// lane = N column (lane&15); half-wave g holds K=16g..16g+15 (32B load).
__device__ inline v16bf load_b(const __bf16* base, int stride, int lane) {
  int n = lane & 15, g = lane >> 4;
  return *reinterpret_cast<const v16bf*>(base + (size_t)n * stride + g * 16);
}

__device__ inline float wave_red_sum(float v) {
#pragma unroll
  for (int m = 16; m; m >>= 1) v += __shfl_xor(v, m, 32);
  return v;
}
__device__ inline float half_red_max(float v) {
#pragma unroll
  for (int m = 8; m; m >>= 1) v = fmaxf(v, __shfl_xor(v, m, 16));
  return v;
}
__device__ inline float half_red_sum(float v) {
#pragma unroll
  for (int m = 8; m; m >>= 1) v += __shfl_xor(v, m, 16);
  return v;
}

// ---------------------------------------------------------------------------
// Tensor Data Mover: 2D tile (bf16) global -> LDS.
// D# packing per CDNA5 ISA:
//  group0: [1:0]=count=1 | [63:32]=lds_addr | [120:64]=global_addr |
//          [127:126]=type=2
//  group1: [17:16]=data_size(1=2B) | [79:48]=tensor_dim0 | [111:80]=tensor_dim1
//          [127:112]=tile_dim0 | [143:128]=tile_dim1 | [207:160]=dim0_stride
// tensor_dim set == tile dims (tiles never go OOB here). Groups 2/3 zero (2D).
// This toolchain's builtin takes 6 args (extra int32x8 group before cpol).
// ---------------------------------------------------------------------------
__device__ inline void tdm_load_2d(const void* gptr, unsigned lds_off,
                                   unsigned tile_x, unsigned tile_y,
                                   unsigned stride_elems) {
  unsigned long long ga = (unsigned long long)(uintptr_t)gptr;
  uint32x4 g0;
  g0[0] = 1u;                                   // count=1, user mode
  g0[1] = lds_off;                              // LDS byte offset
  g0[2] = (unsigned)ga;                         // global addr [95:64]
  g0[3] = (unsigned)((ga >> 32) & 0x1FFFFFFull) // global addr [120:96]
          | (2u << 30);                         // type = 2 ("image")
  int32x8 g1;
  g1[0] = (int)(1u << 16);                      // data_size = 2 bytes
  g1[1] = (int)((tile_x & 0xFFFFu) << 16);      // tensor_dim0 lo16 @ bit48
  g1[2] = (int)((tile_x >> 16) | ((tile_y & 0xFFFFu) << 16)); // td0 hi, td1 lo
  g1[3] = (int)((tile_y >> 16) | (tile_x << 16));             // tile_dim0
  g1[4] = (int)tile_y;                          // tile_dim1 (tile_dim2 = 0)
  g1[5] = (int)stride_elems;                    // tensor_dim0_stride lo32
  g1[6] = 0;
  g1[7] = 0;
  int32x4 z4 = {0, 0, 0, 0};
  int32x8 z8 = {0, 0, 0, 0, 0, 0, 0, 0};
  __builtin_amdgcn_tensor_load_to_lds(g0, g1, z4, z4, z8, 0);
}

__device__ inline unsigned lds_off_of(const void* p) {
  return (unsigned)(uintptr_t)p;   // generic ptr low 32 bits = LDS byte offset
}

// ---------------------------------------------------------------------------
// Weight cast + transpose: W (K x N, f32 row-major) -> Wt (N x K, bf16)
// ---------------------------------------------------------------------------
__global__ __launch_bounds__(256) void castw_kernel(
    const float* __restrict__ W, __bf16* __restrict__ Wt, int K, int N) {
  int idx = blockIdx.x * 256 + threadIdx.x;
  int k = idx / N, n = idx % N;
  Wt[(size_t)n * K + k] = (__bf16)W[idx];
}

// ---------------------------------------------------------------------------
// LayerNorm over E=512 + cast to bf16. One wave per row, 16 elems/lane.
// ---------------------------------------------------------------------------
__global__ __launch_bounds__(256) void ln_cast_kernel(
    const float* __restrict__ x,
    const float* __restrict__ gq, const float* __restrict__ bq,
    const float* __restrict__ gkv, const float* __restrict__ bkv,
    __bf16* __restrict__ qout, __bf16* __restrict__ kvout) {
  int row = blockIdx.x * 8 + (threadIdx.x >> 5);
  int lane = threadIdx.x & 31;
  int n = row & 255;
  int bt = row >> 8;
  int t = bt & 31;
  int b = bt >> 5;

  const float* src = x + (size_t)row * 512 + lane * 16;
  float v[16];
#pragma unroll
  for (int i = 0; i < 16; ++i) v[i] = src[i];
  float s = 0.f, ss = 0.f;
#pragma unroll
  for (int i = 0; i < 16; ++i) { s += v[i]; ss += v[i] * v[i]; }
  s = wave_red_sum(s); ss = wave_red_sum(ss);
  float mean = s * (1.0f / 512.0f);
  float rs = rsqrtf(ss * (1.0f / 512.0f) - mean * mean + 1e-6f);

  const float* g; const float* bb; __bf16* dst;
  if (t == 0) {
    g = gq; bb = bq;
    dst = qout + (size_t)(b * 256 + n) * 512;
  } else {
    g = gkv; bb = bkv;
    dst = kvout + ((size_t)(b * 31 + (t - 1)) * 256 + n) * 512;
  }
#pragma unroll
  for (int i = 0; i < 16; ++i) {
    int e = lane * 16 + i;
    dst[e] = (__bf16)((v[i] - mean) * rs * g[e] + bb[e]);
  }
}

// ---------------------------------------------------------------------------
// res_ln then ln_2 on a precomputed bf16 residual sum; emit bf16.
// ---------------------------------------------------------------------------
__global__ __launch_bounds__(256) void resln2_kernel(
    const __bf16* __restrict__ in,
    const float* __restrict__ g1, const float* __restrict__ b1,
    const float* __restrict__ g2, const float* __restrict__ b2,
    __bf16* __restrict__ out) {
  int row = blockIdx.x * 8 + (threadIdx.x >> 5);
  int lane = threadIdx.x & 31;
  const __bf16* src = in + (size_t)row * 512 + lane * 16;
  float v[16];
#pragma unroll
  for (int i = 0; i < 16; ++i) v[i] = (float)src[i];

  float s = 0.f, ss = 0.f;
#pragma unroll
  for (int i = 0; i < 16; ++i) { s += v[i]; ss += v[i] * v[i]; }
  s = wave_red_sum(s); ss = wave_red_sum(ss);
  float mean = s * (1.0f / 512.0f);
  float rs = rsqrtf(ss * (1.0f / 512.0f) - mean * mean + 1e-6f);
  float y[16];
  float s2 = 0.f, ss2 = 0.f;
#pragma unroll
  for (int i = 0; i < 16; ++i) {
    int e = lane * 16 + i;
    y[i] = (v[i] - mean) * rs * g1[e] + b1[e];
    s2 += y[i]; ss2 += y[i] * y[i];
  }
  s2 = wave_red_sum(s2); ss2 = wave_red_sum(ss2);
  float mean2 = s2 * (1.0f / 512.0f);
  float rs2 = rsqrtf(ss2 * (1.0f / 512.0f) - mean2 * mean2 + 1e-6f);
  __bf16* dst = out + (size_t)row * 512 + lane * 16;
#pragma unroll
  for (int i = 0; i < 16; ++i) {
    int e = lane * 16 + i;
    dst[i] = (__bf16)((y[i] - mean2) * rs2 * g2[e] + b2[e]);
  }
}

// ---------------------------------------------------------------------------
// bf16 WMMA GEMM, TDM-fed and double-buffered:
//   C(MxN) = A(MxK,bf16) @ Wt(NxK,bf16)^T + bias [,GELU] [,+res] -> bf16/f32
// Block tile 128x128xBK32; 256 threads = 8 waves (2x4), 64x32 per wave.
// Wave 0 issues tensor_load_to_lds for tile k+1 while all waves compute
// tile k; s_wait_tensorcnt 2 retires tile k before the barrier.
// Requires M%128==0, N%128==0, K%32==0.
// ---------------------------------------------------------------------------
template <int GELU, int RES, int OUTB>
__global__ __launch_bounds__(256) void gemm_tdm_kernel(
    const __bf16* __restrict__ A, const __bf16* __restrict__ Wt,
    const float* __restrict__ bias, const __bf16* __restrict__ res,
    __bf16* __restrict__ outb, float* __restrict__ outf,
    int M, int N, int K) {
  __shared__ __bf16 As[2][128 * 32];
  __shared__ __bf16 Bs[2][128 * 32];

  int tid = threadIdx.x;
  int lane = tid & 31, w = tid >> 5;
  int wm = w & 1, wn = w >> 1;            // 2 x 4 wave grid
  int bm = blockIdx.x, bn = blockIdx.y;

  v8f acc[4][2];
#pragma unroll
  for (int rt = 0; rt < 4; ++rt)
#pragma unroll
    for (int ct = 0; ct < 2; ++ct) acc[rt][ct] = v8f_zero();

  const __bf16* Ablk = A + (size_t)bm * 128 * K;
  const __bf16* Bblk = Wt + (size_t)bn * 128 * K;
  int ksteps = K >> 5;

  if (tid < 32) {  // wave 0: prime first tiles via TDM
    tdm_load_2d(Ablk, lds_off_of(As[0]), 32u, 128u, (unsigned)K);
    tdm_load_2d(Bblk, lds_off_of(Bs[0]), 32u, 128u, (unsigned)K);
  }

  for (int ks = 0; ks < ksteps; ++ks) {
    int cur = ks & 1;
    if (tid < 32) {
      if (ks + 1 < ksteps) {  // prefetch next tiles into the other buffer
        tdm_load_2d(Ablk + (ks + 1) * 32, lds_off_of(As[cur ^ 1]),
                    32u, 128u, (unsigned)K);
        tdm_load_2d(Bblk + (ks + 1) * 32, lds_off_of(Bs[cur ^ 1]),
                    32u, 128u, (unsigned)K);
        __builtin_amdgcn_s_wait_tensorcnt(2);  // tile ks has landed
      } else {
        __builtin_amdgcn_s_wait_tensorcnt(0);
      }
    }
    __syncthreads();

    AFrag af[4];
#pragma unroll
    for (int rt = 0; rt < 4; ++rt)
      af[rt] = load_a(&As[cur][(wm * 64 + rt * 16) * 32], 32, lane);
    v16bf bf[2];
#pragma unroll
    for (int ct = 0; ct < 2; ++ct)
      bf[ct] = load_b(&Bs[cur][(wn * 32 + ct * 16) * 32], 32, lane);
#pragma unroll
    for (int rt = 0; rt < 4; ++rt)
#pragma unroll
      for (int ct = 0; ct < 2; ++ct)
        acc[rt][ct] = wmma_bf16(af[rt], bf[ct], acc[rt][ct]);
    __syncthreads();  // protect buffer `cur` before it is DMA-overwritten
  }

  // Epilogue (compile-time flags). C layout: lane n = lane&15,
  // row = 8*(lane>>4) + vgpr.
#pragma unroll
  for (int rt = 0; rt < 4; ++rt) {
#pragma unroll
    for (int ct = 0; ct < 2; ++ct) {
      int n_g = bn * 128 + wn * 32 + ct * 16 + (lane & 15);
      float bv = bias[n_g];
#pragma unroll
      for (int rr = 0; rr < 8; ++rr) {
        int m_g = bm * 128 + wm * 64 + rt * 16 + (lane >> 4) * 8 + rr;
        float xv = acc[rt][ct][rr] + bv;
        if (GELU) xv = 0.5f * xv * (1.0f + erff(xv * 0.70710678118654752f));
        if (RES) xv += (float)res[(size_t)m_g * N + n_g];
        if (OUTB) outb[(size_t)m_g * N + n_g] = (__bf16)xv;
        else      outf[(size_t)m_g * N + n_g] = xv;
      }
    }
  }
}

// ---------------------------------------------------------------------------
// Attention: one workgroup per (b,t,h). NF=256, dh=64.
// 512 threads = 16 waves; wave w owns score rows [16w,16w+16).
// Q/K panels staged by TDM; V transposed manually (TDM cannot transpose).
// LDS: Qs 256x64, Ks 256x64, Vt 64x256, Ps 256x256 (bf16) = 224 KB dynamic.
// ---------------------------------------------------------------------------
__global__ __launch_bounds__(512) void attn_kernel(
    const __bf16* __restrict__ qp, const __bf16* __restrict__ kbuf,
    const __bf16* __restrict__ vbuf, __bf16* __restrict__ ctx, float scale) {
  extern __shared__ __bf16 smem[];
  __bf16* Qs = smem;                 // 256*64
  __bf16* Ks = Qs + 16384;           // 256*64
  __bf16* Vt = Ks + 16384;           // 64*256 (d-major)
  __bf16* Ps = Vt + 16384;           // 256*256

  int bx = blockIdx.x;
  int h = bx & 7;
  int bt = bx >> 3;                  // b*31 + t
  int b = bt / 31;

  int tid = threadIdx.x;
  int lane = tid & 31, w = tid >> 5;
  int rw = w * 16;

  if (tid < 32) {  // wave 0: DMA the Q and K panels (64-wide, stride 512)
    tdm_load_2d(qp + ((size_t)b * 256) * 512 + h * 64, lds_off_of(Qs),
                64u, 256u, 512u);
    tdm_load_2d(kbuf + ((size_t)bt * 256) * 512 + h * 64, lds_off_of(Ks),
                64u, 256u, 512u);
  }
  {  // all threads: transpose-stage V while the TDM runs
    int rowi = tid >> 1, hh = tid & 1;
    const __bf16* vs =
        vbuf + ((size_t)bt * 256 + rowi) * 512 + h * 64 + hh * 32;
#pragma unroll
    for (int d = 0; d < 32; ++d) Vt[(hh * 32 + d) * 256 + rowi] = vs[d];
  }
  if (tid < 32) __builtin_amdgcn_s_wait_tensorcnt(0);
  __syncthreads();

  // S = Q @ K^T : 16 rows x 256 cols (16 tiles), K-depth 64 (2 chunks)
  AFrag aq[2];
#pragma unroll
  for (int kc = 0; kc < 2; ++kc)
    aq[kc] = load_a(Qs + rw * 64 + kc * 32, 64, lane);

  v8f s[16];
#pragma unroll
  for (int j = 0; j < 16; ++j) {
    v8f a = v8f_zero();
#pragma unroll
    for (int kc = 0; kc < 2; ++kc) {
      v16bf bk = load_b(Ks + (j * 16) * 64 + kc * 32, 64, lane);
      a = wmma_bf16(aq[kc], bk, a);
    }
    s[j] = a;
  }

  // Softmax (lanes 0-15 rows 0-7, lanes 16-31 rows 8-15 of each 16-row tile)
  float rmax[8], rsum[8];
#pragma unroll
  for (int rr = 0; rr < 8; ++rr) {
    float m = -3.0e38f;
#pragma unroll
    for (int j = 0; j < 16; ++j) m = fmaxf(m, s[j][rr]);
    rmax[rr] = half_red_max(m);
  }
#pragma unroll
  for (int rr = 0; rr < 8; ++rr) {
    int rowm = (lane >> 4) * 8 + rr;
    float sum = 0.0f;
#pragma unroll
    for (int j = 0; j < 16; ++j) {
      float p = expf((s[j][rr] - rmax[rr]) * scale);
      sum += p;
      Ps[(size_t)(rw + rowm) * 256 + j * 16 + (lane & 15)] = (__bf16)p;
    }
    rsum[rr] = half_red_sum(sum);
  }
  __syncthreads();

  // O = P @ V : 16 rows x 64 cols, K-depth 256 (8 chunks)
  v8f o[4];
#pragma unroll
  for (int ct = 0; ct < 4; ++ct) o[ct] = v8f_zero();
#pragma unroll
  for (int kc = 0; kc < 8; ++kc) {
    AFrag ap = load_a(Ps + (size_t)rw * 256 + kc * 32, 256, lane);
#pragma unroll
    for (int ct = 0; ct < 4; ++ct) {
      v16bf bv = load_b(Vt + (size_t)(ct * 16) * 256 + kc * 32, 256, lane);
      o[ct] = wmma_bf16(ap, bv, o[ct]);
    }
  }

  size_t orow0 = (size_t)bt * 256 + rw;
#pragma unroll
  for (int ct = 0; ct < 4; ++ct) {
#pragma unroll
    for (int rr = 0; rr < 8; ++rr) {
      int rowm = (lane >> 4) * 8 + rr;
      float val = o[ct][rr] / rsum[rr];
      ctx[(orow0 + rowm) * 512 + h * 64 + ct * 16 + (lane & 15)] = (__bf16)val;
    }
  }
}

// ---------------------------------------------------------------------------
extern "C" void kernel_launch(void* const* d_in, const int* in_sizes, int n_in,
                              void* d_out, int out_size, void* d_ws,
                              size_t ws_size, hipStream_t stream) {
  (void)in_sizes; (void)n_in; (void)out_size; (void)ws_size;

  constexpr int B = 8, HW = 32, T = 31, NF = 256, E = 512, MH = 2048, H = 8;
  constexpr int RQ = B * NF;           // 2048 query rows
  constexpr int RKV = B * T * NF;      // 63488 kv rows
  constexpr int CH = 7936;             // MLP row chunk (RKV = 8 * 7936)
  constexpr int NCH = RKV / CH;        // 8

  const float* x       = (const float*)d_in[0];
  const float* wq      = (const float*)d_in[1];
  const float* wk      = (const float*)d_in[2];
  const float* wv      = (const float*)d_in[3];
  const float* bq      = (const float*)d_in[4];
  const float* bk      = (const float*)d_in[5];
  const float* bv      = (const float*)d_in[6];
  const float* wo      = (const float*)d_in[7];
  const float* bo      = (const float*)d_in[8];
  const float* lnq_g   = (const float*)d_in[9];
  const float* lnq_b   = (const float*)d_in[10];
  const float* lnkv_g  = (const float*)d_in[11];
  const float* lnkv_b  = (const float*)d_in[12];
  const float* mlpq_w1 = (const float*)d_in[13];
  const float* mlpq_b1 = (const float*)d_in[14];
  const float* mlpq_w2 = (const float*)d_in[15];
  const float* mlpq_b2 = (const float*)d_in[16];
  const float* resln_g = (const float*)d_in[17];
  const float* resln_b = (const float*)d_in[18];
  const float* ln2_g   = (const float*)d_in[19];
  const float* ln2_b   = (const float*)d_in[20];
  const float* mlp_w1  = (const float*)d_in[21];
  const float* mlp_b1  = (const float*)d_in[22];
  const float* mlp_w2  = (const float*)d_in[23];
  const float* mlp_b2  = (const float*)d_in[24];

  // -------- workspace layout (~360 MB) --------
  char* ws = (char*)d_ws;
  size_t off = 0;
  auto take = [&](size_t nbytes) -> char* {
    char* p = ws + off;
    off += (nbytes + 255) & ~(size_t)255;
    return p;
  };
  __bf16* wq_t  = (__bf16*)take((size_t)E * E * 2);
  __bf16* wk_t  = (__bf16*)take((size_t)E * E * 2);
  __bf16* wv_t  = (__bf16*)take((size_t)E * E * 2);
  __bf16* wo_t  = (__bf16*)take((size_t)E * E * 2);
  __bf16* w1q_t = (__bf16*)take((size_t)E * MH * 2);
  __bf16* w2q_t = (__bf16*)take((size_t)MH * E * 2);
  __bf16* w1_t  = (__bf16*)take((size_t)E * MH * 2);
  __bf16* w2_t  = (__bf16*)take((size_t)MH * E * 2);
  __bf16* qln   = (__bf16*)take((size_t)RQ * E * 2);
  __bf16* kvln  = (__bf16*)take((size_t)RKV * E * 2);
  __bf16* qp    = (__bf16*)take((size_t)RQ * E * 2);
  __bf16* kbuf  = (__bf16*)take((size_t)RKV * E * 2);  // later: mlpq_out+res
  __bf16* vbuf  = (__bf16*)take((size_t)RKV * E * 2);  // later: ln2 output
  __bf16* ctx   = (__bf16*)take((size_t)RKV * E * 2);
  __bf16* aproj = (__bf16*)take((size_t)RKV * E * 2);
  __bf16* h1    = (__bf16*)take((size_t)CH * MH * 2);

  // -------- 1) weights -> transposed bf16 --------
  castw_kernel<<<(E * E) / 256, 256, 0, stream>>>(wq, wq_t, E, E);
  castw_kernel<<<(E * E) / 256, 256, 0, stream>>>(wk, wk_t, E, E);
  castw_kernel<<<(E * E) / 256, 256, 0, stream>>>(wv, wv_t, E, E);
  castw_kernel<<<(E * E) / 256, 256, 0, stream>>>(wo, wo_t, E, E);
  castw_kernel<<<(E * MH) / 256, 256, 0, stream>>>(mlpq_w1, w1q_t, E, MH);
  castw_kernel<<<(MH * E) / 256, 256, 0, stream>>>(mlpq_w2, w2q_t, MH, E);
  castw_kernel<<<(E * MH) / 256, 256, 0, stream>>>(mlp_w1, w1_t, E, MH);
  castw_kernel<<<(MH * E) / 256, 256, 0, stream>>>(mlp_w2, w2_t, MH, E);

  // -------- 2) LayerNorms -> bf16 --------
  ln_cast_kernel<<<(B * HW * NF) / 8, 256, 0, stream>>>(
      x, lnq_g, lnq_b, lnkv_g, lnkv_b, qln, kvln);

  // -------- 3) Q/K/V projections --------
  gemm_tdm_kernel<0, 0, 1><<<dim3(RQ / 128, E / 128), 256, 0, stream>>>(
      qln, wq_t, bq, nullptr, qp, nullptr, RQ, E, E);
  gemm_tdm_kernel<0, 0, 1><<<dim3(RKV / 128, E / 128), 256, 0, stream>>>(
      kvln, wk_t, bk, nullptr, kbuf, nullptr, RKV, E, E);
  gemm_tdm_kernel<0, 0, 1><<<dim3(RKV / 128, E / 128), 256, 0, stream>>>(
      kvln, wv_t, bv, nullptr, vbuf, nullptr, RKV, E, E);

  // -------- 4) attention --------
  attn_kernel<<<B * T * H, 512, 229376, stream>>>(qp, kbuf, vbuf, ctx, 0.125f);

  // -------- 5) out projection --------
  gemm_tdm_kernel<0, 0, 1><<<dim3(RKV / 128, E / 128), 256, 0, stream>>>(
      ctx, wo_t, bo, nullptr, aproj, nullptr, RKV, E, E);

  // -------- 6) mlpq + residual (kvln), chunked; sum -> kbuf --------
  for (int c = 0; c < NCH; ++c) {
    const __bf16* ain = aproj + (size_t)c * CH * E;
    gemm_tdm_kernel<1, 0, 1><<<dim3(CH / 128, MH / 128), 256, 0, stream>>>(
        ain, w1q_t, mlpq_b1, nullptr, h1, nullptr, CH, MH, E);
    gemm_tdm_kernel<0, 1, 1><<<dim3(CH / 128, E / 128), 256, 0, stream>>>(
        h1, w2q_t, mlpq_b2, kvln + (size_t)c * CH * E,
        kbuf + (size_t)c * CH * E, nullptr, CH, E, MH);
  }

  // -------- 7) res_ln then ln_2 -> vbuf --------
  resln2_kernel<<<RKV / 8, 256, 0, stream>>>(
      kbuf, resln_g, resln_b, ln2_g, ln2_b, vbuf);

  // -------- 8) final MLP -> d_out (f32), chunked --------
  for (int c = 0; c < NCH; ++c) {
    const __bf16* ain = vbuf + (size_t)c * CH * E;
    gemm_tdm_kernel<1, 0, 1><<<dim3(CH / 128, MH / 128), 256, 0, stream>>>(
        ain, w1_t, mlp_b1, nullptr, h1, nullptr, CH, MH, E);
    gemm_tdm_kernel<0, 0, 0><<<dim3(CH / 128, E / 128), 256, 0, stream>>>(
        h1, w2_t, mlp_b2, nullptr, nullptr,
        (float*)d_out + (size_t)c * CH * E, CH, E, MH);
  }
}
